// MultiHeadAttention_6459630813648
// MI455X (gfx1250) — matrix operand order repaired and checked by reading, not compiled, not run on hardware
//
#include <hip/hip_runtime.h>
#include <hip/hip_bf16.h>

typedef __attribute__((ext_vector_type(16))) _Float16 v16h;
typedef __attribute__((ext_vector_type(8)))  _Float16 v8h;
typedef __attribute__((ext_vector_type(8)))  float    v8f;

#define BN   16
#define HN   16
#define NN   768
#define DD   64
#define EE   1024
#define NEGV (-1e16f)

// ---------------- WMMA helpers (CDNA5 gfx1250, wave32) ----------------

__device__ __forceinline__ v8f wmma_f16(v16h a, v16h b, v8f c) {
    // (neg_a, A, neg_b, B, c_mod, C, reuse_a, reuse_b)
    return __builtin_amdgcn_wmma_f32_16x16x32_f16(false, a, false, b, (short)0, c, false, false);
}

// A fragment: 16x32 f16, row-major source with leading dim ld (elements).
// lane L: M = L%16, half = L/16; elems 0..7 = K[k0+8h .. +7], elems 8..15 = K[k0+16+8h ..]
__device__ __forceinline__ v16h load_frag_a(const _Float16* p, int ld, int row0, int k0) {
    int lane = threadIdx.x & 31;
    int m = lane & 15, half = lane >> 4;
    const _Float16* r = p + (size_t)(row0 + m) * ld + k0 + 8 * half;
    v16h out;
    ((uint4*)&out)[0] = *(const uint4*)(r);
    ((uint4*)&out)[1] = *(const uint4*)(r + 16);
    return out;
}

// B fragment: 32x16 f16 where B[k][n] = Bt[n][k], Bt row-major with ld.
// lane L: n = L%16, half = L/16; elems 0..7 = K[k0+8h .. +7], elems 8..15 = K[k0+16+8h ..]
__device__ __forceinline__ v16h load_frag_b(const _Float16* p, int ld, int n0, int k0) {
    int lane = threadIdx.x & 31;
    int n = lane & 15, half = lane >> 4;
    const _Float16* r = p + (size_t)(n0 + n) * ld + k0 + 8 * half;
    v16h out;
    ((uint4*)&out)[0] = *(const uint4*)(r);
    ((uint4*)&out)[1] = *(const uint4*)(r + 16);
    return out;
}

__device__ __forceinline__ float red_max16(float x) {
    x = fmaxf(x, __shfl_xor(x, 1, 32));
    x = fmaxf(x, __shfl_xor(x, 2, 32));
    x = fmaxf(x, __shfl_xor(x, 4, 32));
    x = fmaxf(x, __shfl_xor(x, 8, 32));
    return x;
}
__device__ __forceinline__ float red_sum16(float x) {
    x += __shfl_xor(x, 1, 32);
    x += __shfl_xor(x, 2, 32);
    x += __shfl_xor(x, 4, 32);
    x += __shfl_xor(x, 8, 32);
    return x;
}

// CDNA5 async global->LDS copy (16 bytes), ASYNCcnt-tracked.
__device__ __forceinline__ void async_copy_b128(_Float16* lds_dst, const _Float16* g_src) {
    unsigned int lds_addr = (unsigned int)(size_t)lds_dst;       // addr[31:0] = LDS offset
    unsigned long long gaddr = (unsigned long long)(size_t)g_src;
    asm volatile("global_load_async_to_lds_b128 %0, %1, off"
                 :: "v"(lds_addr), "v"(gaddr) : "memory");
}
__device__ __forceinline__ void wait_async() {
    asm volatile("s_wait_asynccnt 0x0" ::: "memory");
}

// ---------------- kernel 0: fp32 -> f16 convert ----------------
__global__ void cvt_f32_f16(const float* __restrict__ src, _Float16* __restrict__ dst, int n) {
    int i = blockIdx.x * blockDim.x + threadIdx.x;
    if (i < n) dst[i] = (_Float16)src[i];
}

// ---------------- kernel 1: per-head projection ----------------
// X: (B,N,E) fp32.  W: (H,D,D) fp32 (out = x @ W[h]^T).
// mode 0: out (B*H, N, D) row-major f16   (Q, K)
// mode 1: out (B*H, D, N) transposed f16  (V^T, for PV B-operand)
__global__ void proj_kernel(const float* __restrict__ X, const float* __restrict__ W,
                            _Float16* __restrict__ out, int mode) {
    __shared__ __align__(16) _Float16 Wl[DD * DD];      // 8 KB
    __shared__ __align__(16) _Float16 Xl[4][16 * DD];   // 4 waves * 2 KB

    int bh = blockIdx.x;            // B*H
    int b  = bh >> 4, h = bh & 15;

    for (int i = threadIdx.x; i < DD * DD; i += blockDim.x)
        Wl[i] = (_Float16)W[h * DD * DD + i];
    __syncthreads();

    int wave = threadIdx.x >> 5, lane = threadIdx.x & 31;
    int half = lane >> 4, nn = lane & 15;
    _Float16* xs = Xl[wave];

    for (int t = wave; t < NN / 16; t += 4) {
        int n0 = t * 16;
        const float* src = X + (size_t)b * NN * EE + (size_t)n0 * EE + h * DD;
        for (int i = lane; i < 256; i += 32) {           // 256 float4 chunks
            int rr = i >> 4, c4 = (i & 15) * 4;
            float4 f = *(const float4*)(src + (size_t)rr * EE + c4);
            xs[rr * DD + c4 + 0] = (_Float16)f.x;
            xs[rr * DD + c4 + 1] = (_Float16)f.y;
            xs[rr * DD + c4 + 2] = (_Float16)f.z;
            xs[rr * DD + c4 + 3] = (_Float16)f.w;
        }
        v16h a0 = load_frag_a(xs, DD, 0, 0);
        v16h a1 = load_frag_a(xs, DD, 0, 32);
        #pragma unroll
        for (int et = 0; et < 4; ++et) {
            v8f c = {};
            c = wmma_f16(a0, load_frag_b(Wl, DD, et * 16, 0),  c);
            c = wmma_f16(a1, load_frag_b(Wl, DD, et * 16, 32), c);
            if (mode == 0) {
                _Float16* dst = out + (size_t)bh * NN * DD + (size_t)n0 * DD;
                #pragma unroll
                for (int r = 0; r < 8; ++r)
                    dst[(size_t)(r + 8 * half) * DD + et * 16 + nn] = (_Float16)c[r];
            } else {
                v8h pk;
                #pragma unroll
                for (int r = 0; r < 8; ++r) pk[r] = (_Float16)c[r];
                _Float16* dst = out + (size_t)bh * DD * NN
                              + (size_t)(et * 16 + nn) * NN + n0 + 8 * half;
                *(v8h*)dst = pk;
            }
        }
    }
}

// ---------------- kernel 2: flash attention ----------------
// 4 waves / block, 64 q-rows per block; K/V tiles async-staged to LDS and
// shared by all 4 waves (4x less global K/V traffic).
__global__ void attn_kernel(const _Float16* __restrict__ Qh, const _Float16* __restrict__ Kh,
                            const _Float16* __restrict__ Vt, const float* __restrict__ mask,
                            _Float16* __restrict__ attn) {
    __shared__ __align__(16) _Float16 Ks[32 * DD];       // 32 k-rows x 64 d   (4 KB)
    __shared__ __align__(16) _Float16 Vs[DD * 32];       // 64 d-rows x 32 k   (4 KB)
    __shared__ __align__(16) _Float16 Pl[4][16 * 32];    // per-wave prob tile (4 KB)

    int blk = blockIdx.x;                                 // B*H*(N/64)
    int qblk = blk % (NN / 64);
    int bh   = blk / (NN / 64);
    int b = bh >> 4, h = bh & 15;
    int wave = threadIdx.x >> 5, lane = threadIdx.x & 31;
    int half = lane >> 4, nn = lane & 15;
    int q0 = qblk * 64 + wave * 16;

    const _Float16* Q = Qh + (size_t)bh * NN * DD;
    const _Float16* K = Kh + (size_t)bh * NN * DD;
    const _Float16* V = Vt + (size_t)bh * DD * NN;
    const float* mrow = mask + h * NN;                    // head-indexed mask (reference quirk)

    v16h a0 = load_frag_a(Q, DD, q0, 0);
    v16h a1 = load_frag_a(Q, DD, q0, 32);

    bool bq[8];
    #pragma unroll
    for (int r = 0; r < 8; ++r) bq[r] = (mrow[q0 + 8 * half + r] < -1e30f);

    float m[8], l[8];
    v8f O[4];
    #pragma unroll
    for (int r = 0; r < 8; ++r) { m[r] = -3.0e38f; l[r] = 0.f; }
    #pragma unroll
    for (int dt = 0; dt < 4; ++dt) O[dt] = (v8f){};

    _Float16* Pw = Pl[wave];

    for (int it = 0; it < NN / 32; ++it) {
        int k0 = it * 32;
        __syncthreads();   // all waves done reading previous Ks/Vs
        // stage K tile (32x64) and V^T tile (64x32): 256 16B chunks each, 128 threads
        #pragma unroll
        for (int i = 0; i < 2; ++i) {
            int c = threadIdx.x + i * 128;
            int kr = c >> 3, kc = (c & 7) * 8;            // K: row 0..31, 8-elem chunk
            async_copy_b128(Ks + kr * DD + kc, K + (size_t)(k0 + kr) * DD + kc);
            int vr = c >> 2, vc = (c & 3) * 8;            // V^T: row 0..63, 8-elem chunk
            async_copy_b128(Vs + vr * 32 + vc, V + (size_t)vr * NN + k0 + vc);
        }
        wait_async();
        __syncthreads();   // staged data visible to all waves

        v8f s[2];
        #pragma unroll
        for (int j = 0; j < 2; ++j) {
            v8f c = {};
            c = wmma_f16(a0, load_frag_b(Ks, DD, 16 * j, 0),  c);
            c = wmma_f16(a1, load_frag_b(Ks, DD, 16 * j, 32), c);
            bool bk = (mrow[k0 + 16 * j + nn] < -1e30f);
            #pragma unroll
            for (int r = 0; r < 8; ++r)
                s[j][r] = c[r] * (1.0f / 32.0f) + ((bq[r] || bk) ? NEGV : 0.0f);
        }
        // online softmax row stats (rows live in 16-lane halves)
        #pragma unroll
        for (int r = 0; r < 8; ++r) {
            float mt = red_max16(fmaxf(s[0][r], s[1][r]));
            float mn = fmaxf(m[r], mt);
            float al = __expf(m[r] - mn);
            m[r] = mn;
            float p0 = __expf(s[0][r] - mn);
            float p1 = __expf(s[1][r] - mn);
            s[0][r] = p0; s[1][r] = p1;
            l[r] = l[r] * al + red_sum16(p0 + p1);
            #pragma unroll
            for (int dt = 0; dt < 4; ++dt) O[dt][r] *= al;
        }
        // relayout P: C-fragment -> per-wave LDS -> A-fragment (same-wave, LDS in-order)
        #pragma unroll
        for (int j = 0; j < 2; ++j)
            #pragma unroll
            for (int r = 0; r < 8; ++r)
                Pw[(r + 8 * half) * 32 + j * 16 + nn] = (_Float16)s[j][r];
        __syncthreads();
        v16h pa = load_frag_a(Pw, 32, 0, 0);
        #pragma unroll
        for (int dt = 0; dt < 4; ++dt)
            O[dt] = wmma_f16(pa, load_frag_b(Vs, 32, dt * 16, 0), O[dt]);
    }
    // epilogue: normalize, store as (B,N,H,D) f16
    #pragma unroll
    for (int dt = 0; dt < 4; ++dt) {
        #pragma unroll
        for (int r = 0; r < 8; ++r) {
            size_t q = q0 + r + 8 * half;
            attn[(((size_t)b * NN + q) * HN + h) * DD + dt * 16 + nn] =
                (_Float16)(O[dt][r] / l[r]);
        }
    }
}

// ---------------- kernel 3: out = attn(12288x1024) @ Wf^T ----------------
__global__ void out_gemm(const _Float16* __restrict__ A, const _Float16* __restrict__ Wf16,
                         float* __restrict__ out) {
    int blk = blockIdx.x;                  // (B*N/16) * (E/64)
    int mt = blk / (EE / 64), nt = blk % (EE / 64);
    int m0 = mt * 16, n0 = nt * 64;
    int lane = threadIdx.x & 31, half = lane >> 4, nn = lane & 15;

    v8f C[4];
    #pragma unroll
    for (int dt = 0; dt < 4; ++dt) C[dt] = (v8f){};

    for (int k0 = 0; k0 < EE; k0 += 32) {
        v16h a = load_frag_a(A, EE, m0, k0);
        #pragma unroll
        for (int dt = 0; dt < 4; ++dt)
            C[dt] = wmma_f16(a, load_frag_b(Wf16, EE, n0 + dt * 16, k0), C[dt]);
    }
    #pragma unroll
    for (int dt = 0; dt < 4; ++dt)
        #pragma unroll
        for (int r = 0; r < 8; ++r)
            out[(size_t)(m0 + r + 8 * half) * EE + n0 + dt * 16 + nn] = C[dt][r];
}

// ---------------- host launcher ----------------
extern "C" void kernel_launch(void* const* d_in, const int* in_sizes, int n_in,
                              void* d_out, int out_size, void* d_ws, size_t ws_size,
                              hipStream_t stream) {
    const float* query = (const float*)d_in[0];
    const float* key_t = (const float*)d_in[1];
    const float* value = (const float*)d_in[2];
    const float* mask  = (const float*)d_in[3];
    const float* Wq    = (const float*)d_in[4];
    const float* Wk    = (const float*)d_in[5];
    const float* Wv    = (const float*)d_in[6];
    const float* Wf    = (const float*)d_in[7];
    float* out = (float*)d_out;

    const size_t S1 = (size_t)BN * HN * NN * DD * sizeof(_Float16);  // 25.2 MB
    char* ws = (char*)d_ws;
    _Float16* Qh    = (_Float16*)(ws);
    _Float16* Kh    = (_Float16*)(ws + S1);
    _Float16* Vt    = (_Float16*)(ws + 2 * S1);
    _Float16* attnB = (_Float16*)(ws + 3 * S1);
    _Float16* Wf16  = (_Float16*)(ws + 4 * S1);

    cvt_f32_f16<<<(EE * EE + 255) / 256, 256, 0, stream>>>(Wf, Wf16, EE * EE);

    proj_kernel<<<BN * HN, 128, 0, stream>>>(query, Wq, Qh, 0);
    proj_kernel<<<BN * HN, 128, 0, stream>>>(key_t, Wk, Kh, 0);
    proj_kernel<<<BN * HN, 128, 0, stream>>>(value, Wv, Vt, 1);

    attn_kernel<<<BN * HN * (NN / 64), 128, 0, stream>>>(Qh, Kh, Vt, mask, attnB);

    out_gemm<<<(BN * NN / 16) * (EE / 64), 32, 0, stream>>>(attnB, Wf16, out);
}
